// LoFTREncoderLayer_newcross9_38422777430196
// MI455X (gfx1250) — matrix operand order repaired and checked by reading, not compiled, not run on hardware
//
#include <hip/hip_runtime.h>
#include <hip/hip_bf16.h>
#include <math.h>

typedef __attribute__((ext_vector_type(16))) __bf16 v16bf;
typedef __attribute__((ext_vector_type(8)))  __bf16 v8bf;
typedef __attribute__((ext_vector_type(8)))  float  v8f;
typedef __attribute__((ext_vector_type(4)))  unsigned u32x4;
typedef __attribute__((ext_vector_type(8)))  unsigned u32x8;

// ---------------- problem constants (fixed by reference) ----------------
namespace {
constexpr int kBatch = 8;
constexpr int kH = 80, kW = 80;
constexpr int kHW = 6400;
constexpr int kD = 256;
constexpr int kNi = 3200;               // inside rows per batch (rows 0..39)
constexpr int kPix = kBatch * kHW;      // 51200
constexpr int kMrowsProj = kBatch * kNi;// 25600

constexpr int BLK_M = 128, BLK_N = 64, BLK_K = 32;
constexpr int LDA = 40;                 // padded LDS stride (bf16 elems)
constexpr int LDB = 40;                 // must match TDM pad codes below

// ---------------- workspace layout (bytes) ----------------
constexpr size_t SZ_WT   = (size_t)kD * kD * 2;            // 131072
constexpr size_t OFF_WQT = 0;
constexpr size_t OFF_WKT = OFF_WQT + SZ_WT;
constexpr size_t OFF_WVT = OFF_WKT + SZ_WT;
constexpr size_t SZ_W1B  = (size_t)9 * 256 * 512 * 2;      // 2359296
constexpr size_t OFF_W1B = OFF_WVT + SZ_WT;
constexpr size_t SZ_W2B  = (size_t)9 * 256 * 256 * 2;      // 1179648
constexpr size_t OFF_W2B = OFF_W1B + SZ_W1B;
constexpr size_t OFF_STATS = OFF_W2B + SZ_W2B;             // 1024 floats
constexpr size_t SZ_STATS  = 1024 * 4;
constexpr size_t OFF_Q   = ((OFF_STATS + SZ_STATS + 255) / 256) * 256;
constexpr size_t SZ_QKV  = (size_t)kBatch * kNi * kD * 4;  // 26214400
constexpr size_t OFF_K   = OFF_Q + SZ_QKV;
constexpr size_t OFF_V   = OFF_K + SZ_QKV;
constexpr size_t SZ_KV   = (size_t)kBatch * 8 * 32 * 32 * 4;
constexpr size_t OFF_KV  = OFF_V + SZ_QKV;
constexpr size_t SZ_KS   = (size_t)kBatch * 8 * 32 * 4;
constexpr size_t OFF_KS  = OFF_KV + SZ_KV;
constexpr size_t SZ_X1B  = (size_t)kPix * 512 * 2;         // 52428800
constexpr size_t OFF_X1B = OFF_KS + SZ_KS;
constexpr size_t SZ_Y    = (size_t)kPix * kD * 4;          // 52428800
constexpr size_t OFF_Y1  = OFF_X1B + SZ_X1B;
// reuse: after attention, Q/K/V regions are free
constexpr size_t OFF_X2B = OFF_Q;   // bf16, 26.2 MB fits
constexpr size_t OFF_Y2  = OFF_K;   // f32, spans K+V (52.4 MB)
} // namespace

// ---------------- device helpers ----------------
__device__ __forceinline__ __bf16 f2bf(float f) {
  unsigned u = __builtin_bit_cast(unsigned, f);
  unsigned r = u + 0x7FFFu + ((u >> 16) & 1u);   // round-to-nearest-even
  unsigned short h = (unsigned short)(r >> 16);
  return __builtin_bit_cast(__bf16, h);
}

__device__ __forceinline__ v8f vzero8() {
  v8f z;
#pragma unroll
  for (int i = 0; i < 8; ++i) z[i] = 0.0f;
  return z;
}

// A fragment (16x32 bf16), ISA 7.12.2: lane<16 -> K 0..7,16..23 ; lane>=16 -> K 8..15,24..31
__device__ __forceinline__ v16bf frag_a(const __bf16* sA, int m0, int lane) {
  const int m = m0 + (lane & 15);
  const int off = (lane < 16) ? 0 : 8;
  const __bf16* p = sA + m * LDA + off;
  v8bf lo = *(const v8bf*)p;          // K off..off+7
  v8bf hi = *(const v8bf*)(p + 16);   // K off+16..off+23
  v16bf r;
#pragma unroll
  for (int i = 0; i < 8; ++i) { r[i] = lo[i]; r[i + 8] = hi[i]; }
  return r;
}

// B fragment (32x16 bf16); sB staged transposed [n][k]; lane<16 -> K 0..7,16..23 ; lane>=16 -> K 8..15,24..31
__device__ __forceinline__ v16bf frag_b(const __bf16* sB, int n0, int lane) {
  const int n = n0 + (lane & 15);
  const int kb = (lane < 16) ? 0 : 8;
  const __bf16* p = sB + n * LDB + kb;
  v8bf lo = *(const v8bf*)p;
  v8bf hi = *(const v8bf*)(p + 16);
  v16bf r;
#pragma unroll
  for (int i = 0; i < 8; ++i) { r[i] = lo[i]; r[i + 8] = hi[i]; }
  return r;
}

__device__ __forceinline__ v8f wmma_bf16(v16bf a, v16bf b, v8f c) {
  return __builtin_amdgcn_wmma_f32_16x16x32_bf16(false, a, false, b, (short)0, c,
                                                 false, false);
}

// ---- Tensor Data Mover: 2D bf16 tile (rowElems x rows) global->LDS, with LDS
// row padding reproducing a stride of LDB=40 bf16 (pad_interval code 3 = 16
// DWORDs of data, pad_amount code 3 = 4 DWORDs of pad).  D# per ISA 8.3/8.4.
__device__ __forceinline__ void tdm_load_b_tile(unsigned ldsAddr, const void* gptr,
                                                unsigned rows, unsigned strideElems) {
  unsigned long long ga = (unsigned long long)(size_t)gptr;
  u32x4 g0;
  g0.x = 1u;                                   // count=1 valid descriptor
  g0.y = ldsAddr;                              // lds_addr (bytes)
  g0.z = (unsigned)ga;                         // global_addr[31:0]
  g0.w = ((unsigned)(ga >> 32) & 0x1FFFFFFu)   // global_addr[56:32]
       | (2u << 30);                           // type = 2 ("image")
  u32x8 g1;
#pragma unroll
  for (int i = 0; i < 8; ++i) g1[i] = 0u;
  g1[0] = (1u << 16)                           // data_size = 1 -> 2 bytes
        | (1u << 20)                           // pad_enable
        | (3u << 22)                           // pad_interval: 16 DWORDs
        | (3u << 25);                          // pad_amount:   4 DWORDs
  // tensor_dim0 = tensor_dim1 = 1<<20 (tile always in range)
  g1[1] = 0u;                                  // dim0 low16 << 16 == 0
  g1[2] = 16u;                                 // dim0 hi16 ; dim1 low16<<16 == 0
  g1[3] = 16u | (32u << 16);                   // dim1 hi16 | tile_dim0 = 32
  g1[4] = rows & 0xFFFFu;                      // tile_dim1 ; tile_dim2 = 0
  g1[5] = strideElems;                         // tensor_dim0_stride[31:0]
  asm volatile("tensor_load_to_lds %0, %1" :: "s"(g0), "s"(g1) : "memory");
}

// ---------------- prep kernels ----------------
__global__ void k_cast_bf16(const float* __restrict__ s, __bf16* __restrict__ d, int n) {
  int i = blockIdx.x * blockDim.x + threadIdx.x;
  if (i < n) d[i] = f2bf(s[i]);
}

// OIHW (co,ci,ky,kx) -> [sec=ky*3+kx][co][ci] bf16
__global__ void k_prep_convw(const float* __restrict__ w, __bf16* __restrict__ o,
                             int Cout, int Cin) {
  int n = 9 * Cout * Cin;
  int i = blockIdx.x * blockDim.x + threadIdx.x;
  if (i >= n) return;
  int ci = i % Cin;
  int t = i / Cin;
  int co = t % Cout;
  int sec = t / Cout;
  o[i] = f2bf(w[(size_t)(co * Cin + ci) * 9 + sec]);
}

// feat0 (f32 NHWC) -> x1b channels [0,256); zero tmp-channels [256,512) for inside rows
__global__ void k_cast_feat(const float* __restrict__ feat0, __bf16* __restrict__ x1b) {
  size_t i = (size_t)blockIdx.x * blockDim.x + threadIdx.x;  // over kPix*kD
  int c = (int)(i & (kD - 1));
  size_t pix = i >> 8;
  int s = (int)(pix % kHW);
  x1b[pix * 512 + c] = f2bf(feat0[i]);
  if (s < kNi) x1b[pix * 512 + 256 + c] = f2bf(0.0f);
}

__global__ void k_zero(float* p, int n) {
  int i = blockIdx.x * blockDim.x + threadIdx.x;
  if (i < n) p[i] = 0.0f;
}

// ---------------- projection GEMM (bf16 WMMA) ----------------
// out[b*3200+l, n] = f( sum_k feat0[b, rowOff+l, k] * wT[n, k] )
// MODE 0: elu(x)+1 ; MODE 1: x / Ni
template <int MODE>
__global__ __launch_bounds__(256) void k_gemm_proj(
    const float* __restrict__ feat0, const __bf16* __restrict__ wT,
    float* __restrict__ out, int rowOff) {
  __shared__ __bf16 sA[BLK_M * LDA];
  __shared__ __bf16 sB[BLK_N * LDB];
  const int tid = threadIdx.x, lane = tid & 31, wid = tid >> 5;
  const int wm = wid & 3, wn = wid >> 2;
  const int mBase = blockIdx.x * BLK_M;
  const int nBase = blockIdx.y * BLK_N;

  v8f acc[2][2];
#pragma unroll
  for (int i = 0; i < 2; ++i)
#pragma unroll
    for (int j = 0; j < 2; ++j) acc[i][j] = vzero8();

  const int ar = tid >> 1;             // 0..127 : A tile row
  const int ah = (tid & 1) * 16;       // 0 / 16 : K half
  const int grow = mBase + ar;
  const int bb = grow / kNi;
  const int ss = rowOff + (grow % kNi);
  const float* aSrc = feat0 + ((size_t)bb * kHW + ss) * kD + ah;

  const int brn = tid >> 2;            // 0..63  : B tile n
  const int bc = (tid & 3) * 8;        // 0..24  : K chunk
  const __bf16* bSrcRow = wT + (size_t)(nBase + brn) * kD + bc;

  for (int k0 = 0; k0 < kD; k0 += BLK_K) {
    float tf[16];
    *(float4*)&tf[0]  = *(const float4*)(aSrc + k0 + 0);
    *(float4*)&tf[4]  = *(const float4*)(aSrc + k0 + 4);
    *(float4*)&tf[8]  = *(const float4*)(aSrc + k0 + 8);
    *(float4*)&tf[12] = *(const float4*)(aSrc + k0 + 12);
    __bf16* dstA = &sA[ar * LDA + ah];
#pragma unroll
    for (int j = 0; j < 16; ++j) dstA[j] = f2bf(tf[j]);
    *(uint4*)&sB[brn * LDB + bc] = *(const uint4*)(bSrcRow + k0);
    __syncthreads();

    v16bf aF0 = frag_a(sA, wm * 32 + 0, lane);
    v16bf aF1 = frag_a(sA, wm * 32 + 16, lane);
    v16bf bF0 = frag_b(sB, wn * 32 + 0, lane);
    v16bf bF1 = frag_b(sB, wn * 32 + 16, lane);
    acc[0][0] = wmma_bf16(aF0, bF0, acc[0][0]);
    acc[0][1] = wmma_bf16(aF0, bF1, acc[0][1]);
    acc[1][0] = wmma_bf16(aF1, bF0, acc[1][0]);
    acc[1][1] = wmma_bf16(aF1, bF1, acc[1][1]);
    __syncthreads();
  }

#pragma unroll
  for (int ti = 0; ti < 2; ++ti)
#pragma unroll
    for (int tj = 0; tj < 2; ++tj) {
      const int m0 = mBase + wm * 32 + ti * 16;
      const int n0 = nBase + wn * 32 + tj * 16 + (lane & 15);
#pragma unroll
      for (int e = 0; e < 8; ++e) {
        const int m = m0 + ((lane < 16) ? e : e + 8);
        float vv = acc[ti][tj][e];
        if (MODE == 0) {
          // branchless elu(x)+1 : exp path computed unconditionally, then select
          float eneg = __expf(fminf(vv, 0.0f));
          vv = (vv > 0.0f) ? (vv + 1.0f) : eneg;
        } else {
          vv *= (1.0f / (float)kNi);
        }
        out[(size_t)m * kD + n0] = vv;
      }
    }
}

// ---------------- implicit-GEMM 3x3 conv (bf16 WMMA, TDM-staged weights) ----------------
// out[p, n] = sum_{ky,kx,ci} xin[b, y+ky-1, x+kx-1, ci] * wb[sec][n][ci]
template <int CIN>
__global__ __launch_bounds__(256) void k_gemm_conv(
    const __bf16* __restrict__ xin, const __bf16* __restrict__ wb,
    float* __restrict__ out) {
  __shared__ __bf16 sA[BLK_M * LDA];
  __shared__ __bf16 sB[BLK_N * LDB];
  const int tid = threadIdx.x, lane = tid & 31, wid = tid >> 5;
  const int wm = wid & 3, wn = wid >> 2;
  const int pixBase = blockIdx.x * BLK_M;
  const int nBase = blockIdx.y * BLK_N;
  // wave-uniform wave id for TDM descriptor building
  const int uwid = __builtin_amdgcn_readfirstlane(wid);
  const unsigned sBBase = (unsigned)(size_t)&sB[0];

  v8f acc[2][2];
#pragma unroll
  for (int i = 0; i < 2; ++i)
#pragma unroll
    for (int j = 0; j < 2; ++j) acc[i][j] = vzero8();

  const int ar = tid >> 1;
  const int ah = (tid & 1) * 16;
  const int p = pixBase + ar;
  const int bb = p / kHW;
  const int rem = p % kHW;
  const int py = rem / kW;
  const int px = rem % kW;

  const int Ktot = 9 * CIN;
  for (int kt = 0; kt < Ktot; kt += BLK_K) {
    const int sec = kt / CIN;
    const int ci0 = kt - sec * CIN;
    const int ky = sec / 3, kx = sec - ky * 3;
    const int yy = py + ky - 1, xx = px + kx - 1;
    // per-wave TDM DMA of 8 weight rows (n = nBase+uwid*8 .. +8) x 32 bf16
    {
      const __bf16* gB = wb + ((size_t)(sec * kD + nBase + uwid * 8)) * CIN + ci0;
      tdm_load_b_tile(sBBase + (unsigned)(uwid * 8 * LDB * 2), gB, 8u, (unsigned)CIN);
    }
    // A tile: bounds-checked bf16 rows -> LDS
    uint4 d0 = {0u, 0u, 0u, 0u}, d1 = {0u, 0u, 0u, 0u};
    if (yy >= 0 && yy < kH && xx >= 0 && xx < kW) {
      const __bf16* src = xin + (((size_t)bb * kHW + yy * kW + xx) * CIN + ci0 + ah);
      d0 = *(const uint4*)(src);
      d1 = *(const uint4*)(src + 8);
      __builtin_prefetch(src + 32, 0, 0);   // next K slice (global_prefetch_b8)
    }
    *(uint4*)&sA[ar * LDA + ah + 0] = d0;
    *(uint4*)&sA[ar * LDA + ah + 8] = d1;
    __builtin_amdgcn_s_wait_tensorcnt(0);
    __syncthreads();

    v16bf aF0 = frag_a(sA, wm * 32 + 0, lane);
    v16bf aF1 = frag_a(sA, wm * 32 + 16, lane);
    v16bf bF0 = frag_b(sB, wn * 32 + 0, lane);
    v16bf bF1 = frag_b(sB, wn * 32 + 16, lane);
    acc[0][0] = wmma_bf16(aF0, bF0, acc[0][0]);
    acc[0][1] = wmma_bf16(aF0, bF1, acc[0][1]);
    acc[1][0] = wmma_bf16(aF1, bF0, acc[1][0]);
    acc[1][1] = wmma_bf16(aF1, bF1, acc[1][1]);
    __syncthreads();
  }

#pragma unroll
  for (int ti = 0; ti < 2; ++ti)
#pragma unroll
    for (int tj = 0; tj < 2; ++tj) {
      const int m0 = pixBase + wm * 32 + ti * 16;
      const int n0 = nBase + wn * 32 + tj * 16 + (lane & 15);
#pragma unroll
      for (int e = 0; e < 8; ++e) {
        const int m = m0 + ((lane < 16) ? e : e + 8);
        out[(size_t)m * kD + n0] = acc[ti][tj][e];
      }
    }
}

// ---------------- KV = K^T * (V/Ni), Ksum = sum_s K  (per b,h) ----------------
__global__ __launch_bounds__(256) void k_kv(const float* __restrict__ Kp,
                                            const float* __restrict__ Vp,
                                            float* __restrict__ KVg,
                                            float* __restrict__ Ksg) {
  __shared__ float sK[64][33];
  __shared__ float sV[64][33];
  const int bh = blockIdx.x;          // b*8+h
  const int b = bh >> 3, h = bh & 7;
  const int tid = threadIdx.x;
  const int d = tid >> 3;             // 0..31
  const int v4 = (tid & 7) * 4;       // 0..28
  float a0 = 0.f, a1 = 0.f, a2 = 0.f, a3 = 0.f, ks = 0.f;
  const size_t base = (size_t)b * kNi * kD + h * 32;
  for (int s0 = 0; s0 < kNi; s0 += 64) {
    for (int i = tid; i < 64 * 32; i += 256) {
      int r = i >> 5, c = i & 31;
      size_t g = base + (size_t)(s0 + r) * kD + c;
      sK[r][c] = Kp[g];
      sV[r][c] = Vp[g];
    }
    __syncthreads();
    for (int ssn = 0; ssn < 64; ++ssn) {
      float kd = sK[ssn][d];
      a0 += kd * sV[ssn][v4 + 0];
      a1 += kd * sV[ssn][v4 + 1];
      a2 += kd * sV[ssn][v4 + 2];
      a3 += kd * sV[ssn][v4 + 3];
      if ((tid & 7) == 0) ks += kd;
    }
    __syncthreads();
  }
  float* o = KVg + (size_t)bh * 1024 + d * 32 + v4;
  o[0] = a0; o[1] = a1; o[2] = a2; o[3] = a3;
  if ((tid & 7) == 0) Ksg[(size_t)bh * 32 + d] = ks;
}

// ---------------- message = Ni * Z * (Q @ KV), written bf16 into x1b[.,256:512] ----------------
__global__ __launch_bounds__(256) void k_msg(const float* __restrict__ Qp,
                                             const float* __restrict__ KVg,
                                             const float* __restrict__ Ksg,
                                             __bf16* __restrict__ x1b) {
  __shared__ float sKV[8][32][32];
  __shared__ float sKs[8][32];
  __shared__ float sQ[16][260];
  __shared__ float sZ[16][8];
  const int tid = threadIdx.x;
  const int b = blockIdx.x / 50;
  const int l0 = (blockIdx.x % 50) * 64;
  for (int i = tid; i < 8192; i += 256) ((float*)sKV)[i] = KVg[(size_t)b * 8192 + i];
  if (tid < 256) ((float*)sKs)[tid] = Ksg[(size_t)b * 256 + tid];
  __syncthreads();
  const int h = tid >> 5, v = tid & 31;
  for (int ch = 0; ch < 4; ++ch) {
    const int r0 = l0 + ch * 16;
    for (int i = tid; i < 16 * 256; i += 256) {
      int r = i >> 8, c = i & 255;
      sQ[r][c] = Qp[((size_t)b * kNi + r0 + r) * kD + c];
    }
    __syncthreads();
    if (tid < 128) {
      int r = tid >> 3, hh = tid & 7;
      float dot = 0.f;
#pragma unroll 8
      for (int dd = 0; dd < 32; ++dd) dot += sQ[r][hh * 32 + dd] * sKs[hh][dd];
      sZ[r][hh] = 1.0f / (dot + 1e-6f);
    }
    __syncthreads();
    for (int r = 0; r < 16; ++r) {
      float a = 0.f;
#pragma unroll 8
      for (int dd = 0; dd < 32; ++dd) a += sQ[r][h * 32 + dd] * sKV[h][dd][v];
      float m = a * sZ[r][h] * (float)kNi;
      size_t pix = (size_t)b * kHW + kNi + r0 + r;
      x1b[pix * 512 + 256 + tid] = f2bf(m);
    }
    __syncthreads();
  }
}

// ---------------- batchnorm helpers ----------------
__global__ __launch_bounds__(256) void k_bnstats(const float* __restrict__ y,
                                                 float* __restrict__ sum,
                                                 float* __restrict__ sq) {
  const int c = threadIdx.x;
  float s = 0.f, q = 0.f;
  for (int r = blockIdx.x; r < kPix; r += gridDim.x) {
    float vv = y[(size_t)r * kD + c];
    s += vv;
    q += vv * vv;
  }
  atomicAdd(&sum[c], s);
  atomicAdd(&sq[c], q);
}

__global__ __launch_bounds__(256) void k_bn_cast(const float* __restrict__ y,
                                                 const float* __restrict__ sum,
                                                 const float* __restrict__ sq,
                                                 const float* __restrict__ g,
                                                 const float* __restrict__ be,
                                                 __bf16* __restrict__ ob) {
  const size_t i = (size_t)blockIdx.x * 256 + threadIdx.x;
  const int c = threadIdx.x;
  const float invN = 1.0f / (float)kPix;
  float mean = sum[c] * invN;
  float var = sq[c] * invN - mean * mean;
  float xn = (y[i] - mean) * rsqrtf(var + 1e-5f);
  ob[i] = f2bf(xn * g[c] + be[c]);
}

__global__ __launch_bounds__(256) void k_bn_final(const float* __restrict__ y,
                                                  const float* __restrict__ sum,
                                                  const float* __restrict__ sq,
                                                  const float* __restrict__ g,
                                                  const float* __restrict__ be,
                                                  const float* __restrict__ feat0,
                                                  float* __restrict__ out) {
  const size_t i = (size_t)blockIdx.x * 256 + threadIdx.x;
  const int c = threadIdx.x;
  const float invN = 1.0f / (float)kPix;
  float mean = sum[c] * invN;
  float var = sq[c] * invN - mean * mean;
  float xn = (y[i] - mean) * rsqrtf(var + 1e-5f);
  out[i] = xn * g[c] + be[c] + feat0[i];
}

// ---------------- orchestration ----------------
extern "C" void kernel_launch(void* const* d_in, const int* in_sizes, int n_in,
                              void* d_out, int out_size, void* d_ws, size_t ws_size,
                              hipStream_t stream) {
  (void)in_sizes; (void)n_in; (void)out_size; (void)ws_size;
  const float* feat0   = (const float*)d_in[0];
  const float* w_q     = (const float*)d_in[2];
  const float* w_k     = (const float*)d_in[3];
  const float* w_v     = (const float*)d_in[4];
  const float* conv1_w = (const float*)d_in[5];
  const float* bn1_g   = (const float*)d_in[6];
  const float* bn1_b   = (const float*)d_in[7];
  const float* conv2_w = (const float*)d_in[8];
  const float* bn2_g   = (const float*)d_in[9];
  const float* bn2_b   = (const float*)d_in[10];
  float* out = (float*)d_out;
  char* ws = (char*)d_ws;

  __bf16* wqT = (__bf16*)(ws + OFF_WQT);
  __bf16* wkT = (__bf16*)(ws + OFF_WKT);
  __bf16* wvT = (__bf16*)(ws + OFF_WVT);
  __bf16* w1b = (__bf16*)(ws + OFF_W1B);
  __bf16* w2b = (__bf16*)(ws + OFF_W2B);
  float* sum1 = (float*)(ws + OFF_STATS);
  float* sq1  = sum1 + 256;
  float* sum2 = sum1 + 512;
  float* sq2  = sum1 + 768;
  float* Qb   = (float*)(ws + OFF_Q);
  float* Kb   = (float*)(ws + OFF_K);
  float* Vb   = (float*)(ws + OFF_V);
  float* KVb  = (float*)(ws + OFF_KV);
  float* Ksb  = (float*)(ws + OFF_KS);
  __bf16* x1b = (__bf16*)(ws + OFF_X1B);
  float* y1   = (float*)(ws + OFF_Y1);
  __bf16* x2b = (__bf16*)(ws + OFF_X2B);
  float* y2   = (float*)(ws + OFF_Y2);

  // prep: weight casts/layouts, input cast, stat zeroing
  k_cast_bf16<<<(65536 + 255) / 256, 256, 0, stream>>>(w_q, wqT, 65536);
  k_cast_bf16<<<(65536 + 255) / 256, 256, 0, stream>>>(w_k, wkT, 65536);
  k_cast_bf16<<<(65536 + 255) / 256, 256, 0, stream>>>(w_v, wvT, 65536);
  k_prep_convw<<<(9 * 256 * 512 + 255) / 256, 256, 0, stream>>>(conv1_w, w1b, 256, 512);
  k_prep_convw<<<(9 * 256 * 256 + 255) / 256, 256, 0, stream>>>(conv2_w, w2b, 256, 256);
  k_cast_feat<<<kPix, 256, 0, stream>>>(feat0, x1b);
  k_zero<<<4, 256, 0, stream>>>(sum1, 1024);

  // Q/K/V projections (bf16 WMMA GEMMs)
  dim3 gp(kMrowsProj / BLK_M, kD / BLK_N);   // 200 x 4
  k_gemm_proj<0><<<gp, 256, 0, stream>>>(feat0, wqT, Qb, kNi);  // Q = elu(outside@WqT)+1
  k_gemm_proj<0><<<gp, 256, 0, stream>>>(feat0, wkT, Kb, 0);    // K = elu(inside@WkT)+1
  k_gemm_proj<1><<<gp, 256, 0, stream>>>(feat0, wvT, Vb, 0);    // V = (inside@WvT)/Ni

  // linear attention core
  k_kv<<<kBatch * 8, 256, 0, stream>>>(Kb, Vb, KVb, Ksb);
  k_msg<<<kBatch * 50, 256, 0, stream>>>(Qb, KVb, Ksb, x1b);

  // conv1 (Cin=512) + BN1 -> bf16
  dim3 gc(kPix / BLK_M, kD / BLK_N);         // 400 x 4
  k_gemm_conv<512><<<gc, 256, 0, stream>>>(x1b, w1b, y1);
  k_bnstats<<<256, 256, 0, stream>>>(y1, sum1, sq1);
  k_bn_cast<<<kPix, 256, 0, stream>>>(y1, sum1, sq1, bn1_g, bn1_b, x2b);

  // conv2 (Cin=256) + BN2 + residual
  k_gemm_conv<256><<<gc, 256, 0, stream>>>(x2b, w2b, y2);
  k_bnstats<<<256, 256, 0, stream>>>(y2, sum2, sq2);
  k_bn_final<<<kPix, 256, 0, stream>>>(y2, sum2, sq2, bn2_g, bn2_b, feat0, out);
}
